// VectorQuantizerEMA_56590489092791
// MI455X (gfx1250) — compile-verified, matrix operand order checked by reading, not run on hardware
//
#include <hip/hip_runtime.h>
#include <stdint.h>

// VQ-VAE EMA quantizer for MI455X (gfx1250, wave32).
// vq_main: fused z @ E^T + argmin via split-bf16 WMMA (z=zh+zl, E=eh+el;
//   hi*hi + hi*lo + lo*hi ~= fp32 dot). Full codebook staged in 260 KB LDS.
//   Each wave owns TWO 16-row A tiles (32 rows) -> 12 WMMAs per code tile in
//   4 independent 3-deep chains (hides XDL latency at 2-waves/SIMD occupancy)
//   and halves per-row LDS B traffic. Tile loop unrolled 2x to overlap
//   ds_load_b128 with WMMA issue.
// vq_final: EMA update + normalization (single workgroup).

#define N_TOTAL 131072
#define K_CODES 1024
#define D_DIM   64
#define DECAY   0.99f
#define EPSV    1e-5f

typedef __attribute__((ext_vector_type(16))) __bf16 v16bf;
typedef __attribute__((ext_vector_type(8)))  float  v8f;

__device__ __forceinline__ __bf16 f2bf(float f) {
    uint32_t u = __builtin_bit_cast(uint32_t, f);
    uint32_t r = u + 0x7FFFu + ((u >> 16) & 1u);   // round-to-nearest-even
    uint16_t h = (uint16_t)(r >> 16);
    return __builtin_bit_cast(__bf16, h);
}
__device__ __forceinline__ float bf2f(__bf16 b) {
    uint32_t u = ((uint32_t)__builtin_bit_cast(uint16_t, b)) << 16;
    return __builtin_bit_cast(float, u);
}
__device__ __forceinline__ uint32_t pack2(__bf16 a, __bf16 b) {
    return (uint32_t)__builtin_bit_cast(uint16_t, a) |
           ((uint32_t)__builtin_bit_cast(uint16_t, b) << 16);
}

// Assemble a v16bf A/B operand from two 16-byte LDS chunks (documented 16-bit
// WMMA lane layout: elements 0..7 = K {8h..8h+7}, elements 8..15 = K {16+8h..}).
__device__ __forceinline__ v16bf ld16(const uint32_t* base, int d0, int d1) {
    union { uint4 q[2]; v16bf v; } U;
    U.q[0] = *(const uint4*)(base + d0);
    U.q[1] = *(const uint4*)(base + d1);
    return U.v;
}

// Convert 16 fp32 (already in WMMA element order) into bf16 hi + residual lo.
__device__ __forceinline__ void cvt16(const float* f, v16bf& hi, v16bf& lo) {
    union { uint16_t u[16]; v16bf v; } H, L;
#pragma unroll
    for (int j = 0; j < 16; ++j) {
        __bf16 hb = f2bf(f[j]);
        H.u[j] = __builtin_bit_cast(uint16_t, hb);
        L.u[j] = __builtin_bit_cast(uint16_t, f2bf(f[j] - bf2f(hb)));
    }
    hi = H.v; lo = L.v;
}

// Load this lane's slice (row `zr`) of a 16x64 fp32 A tile and produce the
// four bf16 hi/lo WMMA operands (K chunks 0..31 and 32..63).
__device__ __forceinline__ void load_a_tile(const float* zr, int h,
                                            v16bf& h0, v16bf& l0,
                                            v16bf& h1, v16bf& l1) {
    float fa[16], fb[16];
    float4 a0 = *(const float4*)(zr + 8 * h);
    float4 a1 = *(const float4*)(zr + 8 * h + 4);
    float4 a2 = *(const float4*)(zr + 16 + 8 * h);
    float4 a3 = *(const float4*)(zr + 16 + 8 * h + 4);
    float4 b0 = *(const float4*)(zr + 32 + 8 * h);
    float4 b1 = *(const float4*)(zr + 32 + 8 * h + 4);
    float4 b2 = *(const float4*)(zr + 48 + 8 * h);
    float4 b3 = *(const float4*)(zr + 48 + 8 * h + 4);
    fa[0]=a0.x; fa[1]=a0.y; fa[2]=a0.z; fa[3]=a0.w;
    fa[4]=a1.x; fa[5]=a1.y; fa[6]=a1.z; fa[7]=a1.w;
    fa[8]=a2.x; fa[9]=a2.y; fa[10]=a2.z; fa[11]=a2.w;
    fa[12]=a3.x; fa[13]=a3.y; fa[14]=a3.z; fa[15]=a3.w;
    fb[0]=b0.x; fb[1]=b0.y; fb[2]=b0.z; fb[3]=b0.w;
    fb[4]=b1.x; fb[5]=b1.y; fb[6]=b1.z; fb[7]=b1.w;
    fb[8]=b2.x; fb[9]=b2.y; fb[10]=b2.z; fb[11]=b2.w;
    fb[12]=b3.x; fb[13]=b3.y; fb[14]=b3.z; fb[15]=b3.w;
    cvt16(fa, h0, l0);
    cvt16(fb, h1, l1);
}

__global__ void vq_init_ws(float* ws_counts, float* ws_esum, float* ws_sse) {
    int i = blockIdx.x * blockDim.x + threadIdx.x;
    if (i < K_CODES) ws_counts[i] = 0.f;
    if (i < K_CODES * D_DIM) ws_esum[i] = 0.f;
    if (i == 0) *ws_sse = 0.f;
}

__launch_bounds__(256)
__global__ void vq_main(const float* __restrict__ z,
                        const float* __restrict__ emb,
                        float* __restrict__ out_zq,
                        int*   __restrict__ out_idx,
                        float* __restrict__ ws_counts,
                        float* __restrict__ ws_esum,
                        float* __restrict__ ws_sse) {
    extern __shared__ char smem[];
    uint32_t* ehi = (uint32_t*)smem;              // K x 64 bf16 (32 dwords/row)
    uint32_t* elo = ehi + K_CODES * 32;           // K x 64 bf16
    float*    nrm = (float*)(elo + K_CODES * 32); // K norms ||e_k||^2

    const int tid = threadIdx.x;

    // ---- Stage codebook into LDS as bf16 hi/lo + fp32 norms -----------------
#pragma unroll
    for (int c = 0; c < 4; ++c) {
        int k = tid * 4 + c;
        const float4* row = (const float4*)(emb + (size_t)k * D_DIM);
        float nacc = 0.f;
#pragma unroll
        for (int i = 0; i < 16; ++i) {
            float4 v = row[i];
            nacc += v.x * v.x + v.y * v.y + v.z * v.z + v.w * v.w;
            __bf16 h0 = f2bf(v.x), h1 = f2bf(v.y), h2 = f2bf(v.z), h3 = f2bf(v.w);
            ehi[k * 32 + i * 2 + 0] = pack2(h0, h1);
            ehi[k * 32 + i * 2 + 1] = pack2(h2, h3);
            elo[k * 32 + i * 2 + 0] = pack2(f2bf(v.x - bf2f(h0)), f2bf(v.y - bf2f(h1)));
            elo[k * 32 + i * 2 + 1] = pack2(f2bf(v.z - bf2f(h2)), f2bf(v.w - bf2f(h3)));
        }
        nrm[k] = nacc;
    }
    __syncthreads();

    // ---- Per-wave: two 16-row A tiles (32 rows) ----------------------------
    const int lane = tid & 31;
    const int wave = tid >> 5;
    const int col  = lane & 15;   // M index for A, N index for B, col of C/D
    const int h    = lane >> 4;   // half selector in the 16-bit operand layout
    const int rowBase = (blockIdx.x * 8 + wave) * 32;

    v16bf zh0[2], zl0[2], zh1[2], zl1[2];
#pragma unroll
    for (int r = 0; r < 2; ++r) {
        const float* zr = z + (size_t)(rowBase + r * 16 + col) * D_DIM;
        load_a_tile(zr, h, zh0[r], zl0[r], zh1[r], zl1[r]);
    }

    float bestd[2][8];
    int   bidx[2][8];
#pragma unroll
    for (int r = 0; r < 2; ++r)
#pragma unroll
        for (int i = 0; i < 8; ++i) { bestd[r][i] = 3.4e38f; bidx[r][i] = 0; }

    // ---- Sweep all 1024 codes in 16-code tiles -----------------------------
#pragma unroll 2
    for (int t = 0; t < 64; ++t) {
        int kc = t * 16 + col;
        const uint32_t* bh = ehi + kc * 32;
        const uint32_t* bl = elo + kc * 32;
        v16bf eh0 = ld16(bh, 4 * h,       8 + 4 * h);
        v16bf eh1 = ld16(bh, 16 + 4 * h, 24 + 4 * h);
        v16bf el0 = ld16(bl, 4 * h,       8 + 4 * h);
        v16bf el1 = ld16(bl, 16 + 4 * h, 24 + 4 * h);
        float nc = nrm[kc];

        // 4 independent 3-deep WMMA chains (2 row tiles x 2 K chunks)
        v8f accA[2], accB[2];
        const v8f zero = {0.f, 0.f, 0.f, 0.f, 0.f, 0.f, 0.f, 0.f};
#pragma unroll
        for (int r = 0; r < 2; ++r) {
            accA[r] = __builtin_amdgcn_wmma_f32_16x16x32_bf16(false, zh0[r], false, eh0, (short)0, zero,    false, false);
            accB[r] = __builtin_amdgcn_wmma_f32_16x16x32_bf16(false, zh1[r], false, eh1, (short)0, zero,    false, false);
            accA[r] = __builtin_amdgcn_wmma_f32_16x16x32_bf16(false, zh0[r], false, el0, (short)0, accA[r], false, false);
            accB[r] = __builtin_amdgcn_wmma_f32_16x16x32_bf16(false, zh1[r], false, el1, (short)0, accB[r], false, false);
            accA[r] = __builtin_amdgcn_wmma_f32_16x16x32_bf16(false, zl0[r], false, eh0, (short)0, accA[r], false, false);
            accB[r] = __builtin_amdgcn_wmma_f32_16x16x32_bf16(false, zl1[r], false, eh1, (short)0, accB[r], false, false);
        }

#pragma unroll
        for (int r = 0; r < 2; ++r)
#pragma unroll
            for (int i = 0; i < 8; ++i) {   // d = ||e||^2 - 2 z.e  (||z||^2 dropped)
                float d = fmaf(-2.f, accA[r][i] + accB[r][i], nc);
                if (d < bestd[r][i]) { bestd[r][i] = d; bidx[r][i] = kc; }
            }
    }

    // ---- Cross-lane argmin over the 16 columns (tie -> lowest index) -------
#pragma unroll
    for (int off = 1; off < 16; off <<= 1) {
#pragma unroll
        for (int r = 0; r < 2; ++r)
#pragma unroll
            for (int i = 0; i < 8; ++i) {
                float od = __shfl_xor(bestd[r][i], off, 32);
                int   oi = __shfl_xor(bidx[r][i],  off, 32);
                if (od < bestd[r][i] || (od == bestd[r][i] && oi < bidx[r][i])) {
                    bestd[r][i] = od; bidx[r][i] = oi;
                }
            }
    }

    // ---- Tail: gather z_q, loss, segment-sum scatter -----------------------
    // Row u of tile r lives in vgpr (u&7) of lane 0 (u<8) / lane 16 (u>=8).
    float sse = 0.f;
#pragma unroll
    for (int r = 0; r < 2; ++r) {
#pragma unroll
        for (int u = 0; u < 16; ++u) {
            int k   = __shfl(bidx[r][u & 7], (u & 8) ? 16 : 0, 32);
            int row = rowBase + r * 16 + u;
            float2 zq = *(const float2*)(emb + (size_t)k * D_DIM + 2 * lane);
            float2 zv = *(const float2*)(z   + (size_t)row * D_DIM + 2 * lane);
            *(float2*)(out_zq + (size_t)row * D_DIM + 2 * lane) = zq;
            float dx = zv.x - zq.x, dy = zv.y - zq.y;
            sse += dx * dx + dy * dy;
            atomicAdd(ws_esum + k * D_DIM + 2 * lane,     zv.x);
            atomicAdd(ws_esum + k * D_DIM + 2 * lane + 1, zv.y);
            if (lane == 0) { atomicAdd(ws_counts + k, 1.0f); out_idx[row] = k; }
        }
    }
#pragma unroll
    for (int off = 1; off < 32; off <<= 1) sse += __shfl_xor(sse, off, 32);
    if (lane == 0) atomicAdd(ws_sse, sse);
}

__launch_bounds__(1024)
__global__ void vq_final(const float* __restrict__ cluster_size,
                         const float* __restrict__ ema_embed_sum,
                         const float* __restrict__ ws_counts,
                         const float* __restrict__ ws_esum,
                         const float* __restrict__ ws_sse,
                         float* __restrict__ out_loss,
                         float* __restrict__ out_new_emb,
                         float* __restrict__ out_new_cs,
                         float* __restrict__ out_new_ema) {
    __shared__ float red[K_CODES];
    int k = threadIdx.x;
    float ncs = cluster_size[k] * DECAY + (1.f - DECAY) * ws_counts[k];
    out_new_cs[k] = ncs;
    red[k] = ncs;
    __syncthreads();
    for (int s = K_CODES / 2; s > 0; s >>= 1) {
        if (k < s) red[k] += red[k + s];
        __syncthreads();
    }
    float n   = red[0];
    float cs  = (ncs + EPSV) / (n + (float)K_CODES * EPSV) * n;
    float inv = 1.f / cs;
#pragma unroll 4
    for (int d = 0; d < D_DIM; ++d) {
        float es = ema_embed_sum[k * D_DIM + d] * DECAY +
                   (1.f - DECAY) * ws_esum[k * D_DIM + d];
        out_new_ema[k * D_DIM + d] = es;
        out_new_emb[k * D_DIM + d] = es * inv;
    }
    if (k == 0) *out_loss = *ws_sse / (float)((size_t)N_TOTAL * D_DIM);
}

extern "C" void kernel_launch(void* const* d_in, const int* in_sizes, int n_in,
                              void* d_out, int out_size, void* d_ws, size_t ws_size,
                              hipStream_t stream) {
    const float* z   = (const float*)d_in[0];
    const float* emb = (const float*)d_in[1];
    const float* cs  = (const float*)d_in[2];
    const float* ema = (const float*)d_in[3];

    // Outputs concatenated flat in reference return order.
    float* out      = (float*)d_out;
    float* out_zq   = out;                                         // N*D
    float* out_loss = out + (size_t)N_TOTAL * D_DIM;               // 1
    int*   out_idx  = (int*)(out + (size_t)N_TOTAL * D_DIM + 1);   // N (int32)
    float* out_nemb = out + (size_t)N_TOTAL * D_DIM + 1 + N_TOTAL; // K*D
    float* out_ncs  = out_nemb + K_CODES * D_DIM;                  // K
    float* out_nema = out_ncs + K_CODES;                           // K*D

    float* ws_counts = (float*)d_ws;                  // K
    float* ws_esum   = ws_counts + K_CODES;           // K*D
    float* ws_sse    = ws_esum + K_CODES * D_DIM;     // 1

    vq_init_ws<<<(K_CODES * D_DIM + 255) / 256, 256, 0, stream>>>(ws_counts, ws_esum, ws_sse);

    // 260 KB dynamic LDS: bf16 hi (128K) + bf16 lo (128K) + norms (4K).
    size_t lds_bytes = (size_t)K_CODES * 32 * 4 * 2 + (size_t)K_CODES * 4;
    vq_main<<<N_TOTAL / 256, 256, lds_bytes, stream>>>(
        z, emb, out_zq, out_idx, ws_counts, ws_esum, ws_sse);

    vq_final<<<1, K_CODES, 0, stream>>>(cs, ema, ws_counts, ws_esum, ws_sse,
                                        out_loss, out_nemb, out_ncs, out_nema);
}
